// SphericalCNN_67843303407698
// MI455X (gfx1250) — matrix operand order repaired and checked by reading, not compiled, not run on hardware
//
#include <hip/hip_runtime.h>

typedef __bf16 bf16;
typedef __attribute__((ext_vector_type(16))) __bf16 v16bf;
typedef __attribute__((ext_vector_type(8)))  __bf16 bfx8;
typedef __attribute__((ext_vector_type(8)))  float  v8f;

// ---------- bf16 <-> f32 helpers (round-to-nearest-even) ----------
__device__ __forceinline__ bf16 f2bf(float f) {
  unsigned u = __builtin_bit_cast(unsigned, f);
  unsigned r = u + 0x7FFFu + ((u >> 16) & 1u);
  unsigned short h = (unsigned short)(r >> 16);
  return __builtin_bit_cast(bf16, h);
}
__device__ __forceinline__ float bf2f(bf16 b) {
  unsigned short h = __builtin_bit_cast(unsigned short, b);
  unsigned u = ((unsigned)h) << 16;
  return __builtin_bit_cast(float, u);
}

// ---------- convert fp32 -> bf16 ----------
__global__ void cvt_bf16_kernel(const float* __restrict__ x, bf16* __restrict__ y, size_t n) {
  size_t i = (size_t)blockIdx.x * blockDim.x + threadIdx.x;
  size_t stride = (size_t)gridDim.x * blockDim.x;
  for (; i < n; i += stride) y[i] = f2bf(x[i]);
}

// ---------- weight pre-swizzle into WMMA B layout ----------
// out layout: [nt = Cout/16][t = ceil(K/32)][lane 0..31][16 halves]
//   lane: col = nt*16 + (lane&15);  k = t*32 + (h>=8 ? 16 : 0) + (lane>>4)*8 + (h&7)
__global__ void weight_swizzle_kernel(const float* __restrict__ W, bf16* __restrict__ out,
                                      int K, int Cout, size_t total) {
  int Kt = (K + 31) >> 5;
  size_t i = (size_t)blockIdx.x * blockDim.x + threadIdx.x;
  size_t stride = (size_t)gridDim.x * blockDim.x;
  for (; i < total; i += stride) {
    int h    = (int)(i & 15);
    int lane = (int)((i >> 4) & 31);
    size_t q = i >> 9;
    int t  = (int)(q % (size_t)Kt);
    int nt = (int)(q / (size_t)Kt);
    int k   = t * 32 + ((h >= 8) ? 16 : 0) + (lane >> 4) * 8 + (h & 7);
    int col = nt * 16 + (lane & 15);
    out[i] = (k < K) ? f2bf(W[(size_t)k * Cout + col]) : f2bf(0.0f);
  }
}

// ---------- gather + WMMA conv: z[M,Cout] = gather(act) @ W + bias ----------
// One wave computes a 16 x 32 output tile. A fragments are gathered straight
// into the CDNA5 16-bit A layout: per lane two contiguous 8-half chunks.
__global__ __launch_bounds__(32) void conv_wmma_kernel(
    const bf16* __restrict__ act,    // [B, Nv, Cin] bf16
    const int*  __restrict__ neigh,  // [Nv*7]
    const bf16* __restrict__ wz,     // pre-swizzled [Cout/16][Kt][32][16]
    const float* __restrict__ bias,  // [Cout]
    float* __restrict__ z,           // [B*Nv, Cout] fp32
    int Nv, int Cin, int Cout) {
  const int lane  = threadIdx.x & 31;
  const int mtile = blockIdx.x;
  const int ng    = blockIdx.y;      // group of two 16-col tiles
  const int K  = 7 * Cin;
  const int Kt = (K + 31) >> 5;

  const int m  = mtile * 16 + (lane & 15);
  const int b  = m / Nv;
  const int n  = m - b * Nv;
  const int hi = lane >> 4;
  const int nrow = n * 7;
  const bf16* actb = act + (size_t)b * Nv * Cin;

  const bf16* wz0 = wz + ((size_t)(2 * ng    ) * Kt) * 512;
  const bf16* wz1 = wz + ((size_t)(2 * ng + 1) * Kt) * 512;

  v8f acc0 = {0.f, 0.f, 0.f, 0.f, 0.f, 0.f, 0.f, 0.f};
  v8f acc1 = {0.f, 0.f, 0.f, 0.f, 0.f, 0.f, 0.f, 0.f};
  const bool fast = (Cin & 31) == 0;

  for (int t = 0; t < Kt; ++t) {
    v16bf a;
    if (fast) {
      // per-lane k ranges: [kA, kA+7] and [kA+16, kA+23]; each inside one
      // neighbor row (Cin % 32 == 0 => no j straddle) -> two 16B loads.
      int kA = t * 32 + hi * 8;
      int j0 = kA / Cin;
      int c0 = kA - j0 * Cin;
      int s0 = neigh[nrow + j0];
      bfx8 ch0 = *reinterpret_cast<const bfx8*>(actb + (size_t)s0 * Cin + c0);
      int kB = kA + 16;
      int j1 = kB / Cin;
      int c1 = kB - j1 * Cin;
      int s1 = neigh[nrow + j1];
      bfx8 ch1 = *reinterpret_cast<const bfx8*>(actb + (size_t)s1 * Cin + c1);
#pragma unroll
      for (int i = 0; i < 8; ++i) { a[i] = ch0[i]; a[8 + i] = ch1[i]; }
    } else {
      // generic path (Cin=3, K=21 zero-padded to 32)
#pragma unroll
      for (int h = 0; h < 16; ++h) {
        int k = t * 32 + ((h >= 8) ? 16 : 0) + hi * 8 + (h & 7);
        bf16 val = f2bf(0.0f);
        if (k < K) {
          int j = k / Cin;
          int c = k - j * Cin;
          int s = neigh[nrow + j];
          val = actb[(size_t)s * Cin + c];
        }
        a[h] = val;
      }
    }
    v16bf bb0 = *reinterpret_cast<const v16bf*>(wz0 + ((size_t)t * 32 + lane) * 16);
    v16bf bb1 = *reinterpret_cast<const v16bf*>(wz1 + ((size_t)t * 32 + lane) * 16);

    acc0 = __builtin_amdgcn_wmma_f32_16x16x32_bf16(false, a, false, bb0, (short)0, acc0, false, false);
    acc1 = __builtin_amdgcn_wmma_f32_16x16x32_bf16(false, a, false, bb1, (short)0, acc1, false, false);
  }

  const int col0 = ng * 32 + (lane & 15);
  const float b0 = bias[col0];
  const float b1 = bias[col0 + 16];
  const int rbase = mtile * 16 + hi * 8;
#pragma unroll
  for (int v = 0; v < 8; ++v) {
    size_t r = (size_t)(rbase + v) * Cout;
    z[r + col0]      = acc0[v] + b0;
    z[r + col0 + 16] = acc1[v] + b1;
  }
}

// ---------- deterministic per-channel sum / sumsq partials ----------
__global__ __launch_bounds__(256) void stats_kernel(const float* __restrict__ z,
                                                    float* __restrict__ pbuf, int M, int C) {
  __shared__ float ssum[256];
  __shared__ float ssq[256];
  int t = threadIdx.x;
  int c  = t & (C - 1);            // C is a power of two (32..256)
  int rg = t / C;
  int rstep = blockDim.x / C;
  float s0 = 0.f, s1 = 0.f;
  for (int r = blockIdx.x * rstep + rg; r < M; r += gridDim.x * rstep) {
    float v = z[(size_t)r * C + c];
    s0 += v; s1 += v * v;
  }
  ssum[t] = s0; ssq[t] = s1;
  __syncthreads();
  for (int s = 128; s >= C; s >>= 1) {
    if (t < s) { ssum[t] += ssum[t + s]; ssq[t] += ssq[t + s]; }
    __syncthreads();
  }
  if (t < C) {
    pbuf[(size_t)blockIdx.x * 512 + t]       = ssum[t];
    pbuf[(size_t)blockIdx.x * 512 + 256 + t] = ssq[t];
  }
}

// ---------- fold partials -> per-channel BN scale/shift ----------
__global__ void finalize_kernel(const float* __restrict__ pbuf, int nblocks,
                                const float* __restrict__ gamma, const float* __restrict__ beta,
                                float* __restrict__ ss, int M, int C) {
  int c = threadIdx.x;
  if (c >= C) return;
  float s0 = 0.f, s1 = 0.f;
  for (int i = 0; i < nblocks; ++i) {
    s0 += pbuf[(size_t)i * 512 + c];
    s1 += pbuf[(size_t)i * 512 + 256 + c];
  }
  float inv  = 1.0f / (float)M;
  float mean = s0 * inv;
  float var  = s1 * inv - mean * mean;
  float sc   = gamma[c] * rsqrtf(var + 1e-5f);
  ss[c]       = sc;
  ss[256 + c] = beta[c] - mean * sc;
}

// ---------- fused BN + LeakyReLU, fp32 -> bf16 ----------
__global__ void bn_lrelu_kernel(const float* __restrict__ z, const float* __restrict__ ss,
                                bf16* __restrict__ y, size_t total, int C) {
  size_t i = (size_t)blockIdx.x * blockDim.x + threadIdx.x;
  size_t stride = (size_t)gridDim.x * blockDim.x;
  for (; i < total; i += stride) {
    int c = (int)(i % (size_t)C);
    float v = z[i] * ss[c] + ss[256 + c];
    v = (v >= 0.0f) ? v : 0.2f * v;
    y[i] = f2bf(v);
  }
}

// ---------- mean pool over 1-ring of fine mesh ----------
__global__ void pool_kernel(const bf16* __restrict__ in, const int* __restrict__ neigh,
                            bf16* __restrict__ out, int NvIn, int NvOut, int C, size_t total) {
  size_t i = (size_t)blockIdx.x * blockDim.x + threadIdx.x;
  size_t stride = (size_t)gridDim.x * blockDim.x;
  for (; i < total; i += stride) {
    int c = (int)(i % (size_t)C);
    size_t t2 = i / (size_t)C;
    int nO = (int)(t2 % (size_t)NvOut);
    int b  = (int)(t2 / (size_t)NvOut);
    const bf16* inb = in + (size_t)b * NvIn * C;
    float s = 0.f;
#pragma unroll
    for (int j = 0; j < 7; ++j)
      s += bf2f(inb[(size_t)neigh[nO * 7 + j] * C + c]);
    out[i] = f2bf(s * (1.0f / 7.0f));
  }
}

// ---------- Conv1d(256,1,k=1): per-vertex dot ----------
__global__ void head_kernel(const bf16* __restrict__ act, const float* __restrict__ cw,
                            const float* __restrict__ cb, float* __restrict__ ycat,
                            int Nv, int C, int B, int colOff) {
  int idx = blockIdx.x * blockDim.x + threadIdx.x;
  if (idx >= B * Nv) return;
  int b = idx / Nv, nn = idx - b * Nv;
  const bf16* p = act + (size_t)idx * C;
  float s = 0.f;
  for (int c = 0; c < C; ++c) s += bf2f(p[c]) * cw[c];
  ycat[(size_t)b * 324 + colOff + nn] = s + cb[0];
}

// ---------- Linear(324,2) ----------
__global__ void fc_kernel(const float* __restrict__ ycat, const float* __restrict__ fcW,
                          const float* __restrict__ fcb, float* __restrict__ out, int B) {
  int idx = blockIdx.x * blockDim.x + threadIdx.x;
  if (idx >= B * 2) return;
  int b = idx >> 1, k = idx & 1;
  float s = fcb[k];
  const float* row = ycat + (size_t)b * 324;
  for (int j = 0; j < 324; ++j) s += row[j] * fcW[j * 2 + k];
  out[idx] = s;
}

extern "C" void kernel_launch(void* const* d_in, const int* in_sizes, int n_in,
                              void* d_out, int out_size, void* d_ws, size_t ws_size,
                              hipStream_t stream) {
  (void)n_in; (void)out_size; (void)ws_size;
  const float* x[2] = {(const float*)d_in[0], (const float*)d_in[1]};
  const int* neighs[4] = {(const int*)d_in[2], (const int*)d_in[3],
                          (const int*)d_in[4], (const int*)d_in[5]};
  const int levels[4] = {10242, 2562, 642, 162};
  const int cins[4]   = {3, 32, 64, 128};
  const int couts[4]  = {32, 64, 128, 256};
  const int B = in_sizes[0] / (10242 * 3);   // 32

  // ---- carve workspace ----
  char* base = (char*)d_ws;
  size_t off = 0;
  auto carve = [&](size_t bytes) -> char* {
    char* p = base + off;
    off += (bytes + 255) & ~(size_t)255;
    return p;
  };
  size_t maxElems = (size_t)B * 10242 * 32;          // largest activation/z
  float* zbuf = (float*)carve(maxElems * 4);
  bf16* actA  = (bf16*)carve(maxElems * 2);
  bf16* actB  = (bf16*)carve(maxElems * 2);
  size_t woffs[8]; size_t wtot = 0;
  for (int blk = 0; blk < 4; ++blk)
    for (int tag = 0; tag < 2; ++tag) {
      int Cin = tag ? couts[blk] : cins[blk];
      int K = 7 * Cin, Kt = (K + 31) >> 5;
      woffs[blk * 2 + tag] = wtot;
      wtot += (size_t)(couts[blk] / 16) * Kt * 512;
    }
  bf16* wzbase = (bf16*)carve(wtot * 2);
  const int STATS_BLOCKS = 128;
  float* pbuf  = (float*)carve((size_t)STATS_BLOCKS * 512 * 4);
  float* ssbuf = (float*)carve(512 * 4);
  float* ycat  = (float*)carve((size_t)B * 324 * 4);

  // ---- pre-swizzle all 8 weight matrices to bf16 WMMA-B layout ----
  for (int blk = 0; blk < 4; ++blk)
    for (int tag = 0; tag < 2; ++tag) {
      int Cin = tag ? couts[blk] : cins[blk];
      int Cout = couts[blk];
      int K = 7 * Cin, Kt = (K + 31) >> 5;
      size_t total = (size_t)(Cout / 16) * Kt * 512;
      const float* W = (const float*)d_in[6 + blk * 8 + tag * 4];
      int blocks = (int)((total + 255) / 256);
      weight_swizzle_kernel<<<blocks, 256, 0, stream>>>(W, wzbase + woffs[blk * 2 + tag],
                                                        K, Cout, total);
    }

  // ---- two branches ----
  for (int side = 0; side < 2; ++side) {
    size_t nx = (size_t)B * 10242 * 3;
    cvt_bf16_kernel<<<(int)((nx + 255) / 256), 256, 0, stream>>>(x[side], actA, nx);
    bf16* cur = actA;
    bf16* alt = actB;
    for (int blk = 0; blk < 4; ++blk) {
      int Nv = levels[blk];
      int M = B * Nv;
      for (int tag = 0; tag < 2; ++tag) {
        int Cin  = tag ? couts[blk] : cins[blk];
        int Cout = couts[blk];
        const float* bias  = (const float*)d_in[6 + blk * 8 + tag * 4 + 1];
        const float* gamma = (const float*)d_in[6 + blk * 8 + tag * 4 + 2];
        const float* beta  = (const float*)d_in[6 + blk * 8 + tag * 4 + 3];
        dim3 grid(M / 16, Cout / 32);
        conv_wmma_kernel<<<grid, 32, 0, stream>>>(cur, neighs[blk],
                                                  wzbase + woffs[blk * 2 + tag],
                                                  bias, zbuf, Nv, Cin, Cout);
        stats_kernel<<<STATS_BLOCKS, 256, 0, stream>>>(zbuf, pbuf, M, Cout);
        finalize_kernel<<<1, 256, 0, stream>>>(pbuf, STATS_BLOCKS, gamma, beta, ssbuf, M, Cout);
        size_t tot = (size_t)M * Cout;
        bn_lrelu_kernel<<<1024, 256, 0, stream>>>(zbuf, ssbuf, alt, tot, Cout);
        bf16* tmp = cur; cur = alt; alt = tmp;
      }
      if (blk < 3) {
        int NvOut = levels[blk + 1];
        size_t tot = (size_t)B * NvOut * couts[blk];
        pool_kernel<<<(int)((tot + 255) / 256), 256, 0, stream>>>(cur, neighs[blk], alt,
                                                                  Nv, NvOut, couts[blk], tot);
        bf16* tmp = cur; cur = alt; alt = tmp;
      }
    }
    const float* cw = (const float*)d_in[38];
    const float* cb = (const float*)d_in[39];
    int tot = B * 162;
    head_kernel<<<(tot + 127) / 128, 128, 0, stream>>>(cur, cw, cb, ycat, 162, 256, B, side * 162);
  }

  const float* fcW = (const float*)d_in[40];
  const float* fcb = (const float*)d_in[41];
  fc_kernel<<<1, 64, 0, stream>>>(ycat, fcW, fcb, (float*)d_out, B);
}
